// SRUPlusPlus_37288906064642
// MI455X (gfx1250) — matrix operand order repaired and checked by reading, not compiled, not run on hardware
//
#include <hip/hip_runtime.h>
#include <hip/hip_bf16.h>
#include <stdint.h>

// ---------------------------------------------------------------------------
// Types for CDNA5 WMMA (wave32): 16x16x32 bf16 -> f32
// ---------------------------------------------------------------------------
typedef __bf16        bf16_t;
typedef __attribute__((ext_vector_type(16))) __bf16        v16bf;
typedef __attribute__((ext_vector_type(8)))  float         v8f;
typedef __attribute__((ext_vector_type(8)))  unsigned int  v8u;

// Pointer types for the async global->LDS builtin (16B granule = int4)
typedef int v4i_ __attribute__((vector_size(16)));
typedef __attribute__((address_space(1))) v4i_* p_glb_v4i;   // global (AS1)
typedef __attribute__((address_space(3))) v4i_* p_lds_v4i;   // LDS (AS3)

// Problem constants (B,N,D,H) = (8,2048,1024,1024)
#define PB   8
#define PN   2048
#define PD   1024
#define PH   1024
#define PH3  3072
#define PBN  (PB * PN)      // 16384

// GEMM tiling
#define BM  128
#define BNT 128
#define BK  32
#define LDT 40              // padded LDS row stride (elements); 80B rows, 16B-aligned chunks

// ---------------------------------------------------------------------------
// CDNA5 async global->LDS copy (ASYNCcnt-tracked), with sync fallback.
// ---------------------------------------------------------------------------
#if defined(__gfx1250__) && \
    __has_builtin(__builtin_amdgcn_global_load_async_to_lds_b128) && \
    __has_builtin(__builtin_amdgcn_s_wait_asynccnt)
#define USE_ASYNC_LDS 1
#else
#define USE_ASYNC_LDS 0
#endif

__device__ __forceinline__ void cp16_to_lds(const bf16_t* g, bf16_t* l)
{
#if USE_ASYNC_LDS
    __builtin_amdgcn_global_load_async_to_lds_b128(
        (p_glb_v4i)(uintptr_t)g,
        (p_lds_v4i)(uintptr_t)(unsigned int)(uintptr_t)l,
        0, 0);
#else
    *(uint4*)l = *(const uint4*)g;
#endif
}

__device__ __forceinline__ void wait_async_copies()
{
#if USE_ASYNC_LDS
    __builtin_amdgcn_s_wait_asynccnt(0);
#endif
}

// ---------------------------------------------------------------------------
// Fragment loads from LDS following the ISA wave32 VGPR layouts.
// A (16x32 bf16, MxK): VGPR v holds K = (v>=4?16:0) + half*8 + 2*(v&3), pairs.
// B (32x16 bf16, KxN) stored n-major (Bt[n][k]): VGPR v holds K = (v>=4?16:0) + half*8 + 2*(v&3).
// ---------------------------------------------------------------------------
__device__ __forceinline__ v16bf load_frag_a(const bf16_t* As, int row0, int lane) {
    const int m    = lane & 15;
    const int half = (lane >> 4) & 1;
    const char* p = (const char*)(As + (size_t)(row0 + m) * LDT);
    v8u u;
#pragma unroll
    for (int v = 0; v < 8; ++v) {
        const int k = ((v & 4) << 2) + half * 8 + 2 * (v & 3);
        u[v] = *(const unsigned int*)(p + 2 * k);
    }
    return __builtin_bit_cast(v16bf, u);
}

__device__ __forceinline__ v16bf load_frag_b(const bf16_t* Bt, int col0, int lane) {
    const int n    = lane & 15;
    const int half = (lane >> 4) & 1;
    const char* p = (const char*)(Bt + (size_t)(col0 + n) * LDT);
    v8u u;
#pragma unroll
    for (int v = 0; v < 8; ++v) {
        const int k = ((v & 4) << 2) + half * 8 + 2 * (v & 3);
        u[v] = *(const unsigned int*)(p + 2 * k);
    }
    return __builtin_bit_cast(v16bf, u);
}

// ---------------------------------------------------------------------------
// bf16 WMMA GEMM, B always n-major:  C[M,Nn] = scale * (A[M,:K] @ Bg[Nn,:K]^T) (+bias)
//   ldA/ldB/ldC: row strides (elements).  OUT_TRANS: store C column-major (C^T),
//   i.e. off = gcol*ldC + grow (used to produce V^T directly).
// Batched via blockIdx.z with element strides sA, sB, sC.
// Block: 256 threads = 8 waves (2x4); wave computes 64x32 = 8 WMMA tiles.
// Double-buffered LDS; staging via async global->LDS b128 copies.
// ---------------------------------------------------------------------------
template <bool OUT_BF16, bool OUT_TRANS, bool HAS_BIAS>
__global__ void __launch_bounds__(256)
gemm_tn(const bf16_t* __restrict__ A, const bf16_t* __restrict__ Bg,
        void* __restrict__ Cout, const float* __restrict__ bias,
        int M, int Nn, int K, int ldA, int ldB, int ldC,
        long long sA, long long sB, long long sC, float scale)
{
    __shared__ bf16_t As[2][BM * LDT];
    __shared__ bf16_t Bt[2][BNT * LDT];

    const int tid   = threadIdx.x;
    const int lane  = tid & 31;
    const int wave  = tid >> 5;
    const int waveM = wave >> 2;    // 0..1
    const int waveN = wave & 3;     // 0..3

    const int rowBase = blockIdx.x * BM;
    const int colBase = blockIdx.y * BNT;
    const long long bz = blockIdx.z;
    A  += bz * sA;
    Bg += bz * sB;

    v8f acc[4][2];
#pragma unroll
    for (int i = 0; i < 4; ++i)
#pragma unroll
        for (int j = 0; j < 2; ++j) acc[i][j] = (v8f){0,0,0,0,0,0,0,0};

    const int r  = tid >> 2;            // 0..63
    const int c8 = (tid & 3) * 8;       // 0,8,16,24

    auto stage = [&](int buf, int kt) {
#pragma unroll
        for (int rr = 0; rr < BM; rr += 64) {
            cp16_to_lds(A  + (size_t)(rowBase + r + rr) * ldA + kt * BK + c8,
                        &As[buf][(r + rr) * LDT + c8]);
            cp16_to_lds(Bg + (size_t)(colBase + r + rr) * ldB + kt * BK + c8,
                        &Bt[buf][(r + rr) * LDT + c8]);
        }
    };

    const int nK = K / BK;
    stage(0, 0);
    int buf = 0;

    for (int kt = 0; kt < nK; ++kt) {
        wait_async_copies();            // tile kt landed in LDS (this wave's copies)
        __syncthreads();                // all waves' copies done; prev compute done
        if (kt + 1 < nK) stage(buf ^ 1, kt + 1);

        v16bf af[4];
#pragma unroll
        for (int i = 0; i < 4; ++i)
            af[i] = load_frag_a(As[buf], waveM * 64 + i * 16, lane);
        v16bf bfg[2];
#pragma unroll
        for (int j = 0; j < 2; ++j)
            bfg[j] = load_frag_b(Bt[buf], waveN * 32 + j * 16, lane);

#pragma unroll
        for (int i = 0; i < 4; ++i)
#pragma unroll
            for (int j = 0; j < 2; ++j)
                acc[i][j] = __builtin_amdgcn_wmma_f32_16x16x32_bf16(
                    false, af[i], false, bfg[j], (short)0, acc[i][j], false, false);

        buf ^= 1;
    }

    // ---- epilogue: C layout (VGPRr: lanes0-15 M=r, lanes16-31 M=8+r; N=lane&15) ----
    const int nIn = lane & 15;
    const int mHi = (lane >> 4) * 8;
    const long long cBase = bz * sC;
#pragma unroll
    for (int i = 0; i < 4; ++i) {
#pragma unroll
        for (int j = 0; j < 2; ++j) {
            const int gcol = colBase + waveN * 32 + j * 16 + nIn;
            const float badd = HAS_BIAS ? bias[gcol] : 0.0f;
#pragma unroll
            for (int rr = 0; rr < 8; ++rr) {
                const int grow = rowBase + waveM * 64 + i * 16 + mHi + rr;
                const float v = acc[i][j][rr] * scale + badd;
                const long long off = OUT_TRANS
                    ? cBase + (long long)gcol * ldC + grow
                    : cBase + (long long)grow * ldC + gcol;
                if (OUT_BF16) ((bf16_t*)Cout)[off] = (bf16_t)v;
                else          ((float*)Cout)[off]  = v;
            }
        }
    }
}

// ---------------------------------------------------------------------------
// Elementwise / convert helpers
// ---------------------------------------------------------------------------
__global__ void __launch_bounds__(256)
f32_to_bf16_kernel(const float* __restrict__ in, bf16_t* __restrict__ out, size_t n)
{
    for (size_t i = (size_t)blockIdx.x * 256 + threadIdx.x; i < n;
         i += (size_t)gridDim.x * 256)
        out[i] = (bf16_t)in[i];
}

// in: R x C row-major fp32 -> out: C x R row-major bf16 (transposed convert)
__global__ void __launch_bounds__(256)
f32_to_bf16_t_kernel(const float* __restrict__ in, bf16_t* __restrict__ out, int R, int C)
{
    __shared__ float t[32][33];
    const int tx = threadIdx.x & 31;
    const int ty = threadIdx.x >> 5;        // 0..7
    const int rb = blockIdx.y * 32, cb = blockIdx.x * 32;
#pragma unroll
    for (int i = 0; i < 32; i += 8)
        t[ty + i][tx] = in[(size_t)(rb + ty + i) * C + cb + tx];
    __syncthreads();
#pragma unroll
    for (int i = 0; i < 32; i += 8)
        out[(size_t)(cb + ty + i) * R + rb + tx] = (bf16_t)t[tx][ty + i];
}

__global__ void __launch_bounds__(256)
make_g_kernel(const float* __restrict__ Q, const float* __restrict__ Fa,
              const float* __restrict__ alpha, bf16_t* __restrict__ G, size_t n)
{
    const float a = alpha[0];
    for (size_t i = (size_t)blockIdx.x * 256 + threadIdx.x; i < n;
         i += (size_t)gridDim.x * 256)
        G[i] = (bf16_t)(Q[i] + a * Fa[i]);
}

// Row softmax over rows of length n (==2048); one 256-thread block per row; bf16 out
__global__ void __launch_bounds__(256)
softmax_rows_kernel(const float* __restrict__ S, bf16_t* __restrict__ P, int n)
{
    __shared__ float red[256];
    const int tid = threadIdx.x;
    const size_t base = (size_t)blockIdx.x * n;
    const float* s = S + base;
    bf16_t* p = P + base;

    float m = -3.402823466e38f;
    for (int i = tid; i < n; i += 256) m = fmaxf(m, s[i]);
    red[tid] = m; __syncthreads();
    for (int o = 128; o > 0; o >>= 1) {
        if (tid < o) red[tid] = fmaxf(red[tid], red[tid + o]);
        __syncthreads();
    }
    m = red[0]; __syncthreads();

    float sum = 0.0f;
    for (int i = tid; i < n; i += 256) sum += __expf(s[i] - m);
    red[tid] = sum; __syncthreads();
    for (int o = 128; o > 0; o >>= 1) {
        if (tid < o) red[tid] += red[tid + o];
        __syncthreads();
    }
    const float inv = 1.0f / red[0];
    for (int i = tid; i < n; i += 256)
        p[i] = (bf16_t)(__expf(s[i] - m) * inv);
}

// ---------------------------------------------------------------------------
// SRU scan: one thread per (b,h); recurrence independent across (b,h).
// ---------------------------------------------------------------------------
__global__ void __launch_bounds__(256)
sru_scan_kernel(const float* __restrict__ U, const float* __restrict__ x,
                const float* __restrict__ vf, const float* __restrict__ vr,
                const float* __restrict__ bfv, const float* __restrict__ brv,
                float* __restrict__ Hout)
{
    const int idx = blockIdx.x * 256 + threadIdx.x;   // 0 .. B*H-1
    const int b = idx / PH;
    const int h = idx % PH;
    const float vfh = vf[h], vrh = vr[h], bfh = bfv[h], brh = brv[h];
    float c = 0.0f;

    for (int t = 0; t < PN; ++t) {
        const size_t rbase = ((size_t)b * PN + t);
        const size_t ub = rbase * PH3;
        if (t + 1 < PN)
            __builtin_prefetch(U + ub + PH3 + h, 0, 1);   // global_prefetch_b8
        const float u0 = U[ub + h];
        const float u1 = U[ub + PH + h];
        const float u2 = U[ub + 2 * PH + h];
        const float xt = x[rbase * PH + h];

        const float f  = 1.0f / (1.0f + __expf(-(u0 + vfh * c + bfh)));
        const float cn = f * c + (1.0f - f) * u2;
        const float rg = 1.0f / (1.0f + __expf(-(u1 + vrh * c + brh)));
        Hout[rbase * PH + h] = rg * cn + (1.0f - rg) * xt;
        c = cn;
    }
}

// ---------------------------------------------------------------------------
// Launch orchestration
// ---------------------------------------------------------------------------
extern "C" void kernel_launch(void* const* d_in, const int* in_sizes, int n_in,
                              void* d_out, int out_size, void* d_ws, size_t ws_size,
                              hipStream_t stream)
{
    const float* x_f  = (const float*)d_in[0];   // (B,N,D)
    const float* W_f  = (const float*)d_in[1];   // (3H,D)
    const float* Wq_f = (const float*)d_in[2];   // (D,H)
    const float* Wk_f = (const float*)d_in[3];   // (H,H)
    const float* Wv_f = (const float*)d_in[4];   // (H,H)
    const float* Wu_f = (const float*)d_in[5];   // (H,3H)
    const float* bu_f = (const float*)d_in[6];   // (3H,)
    const float* vf_f = (const float*)d_in[7];
    const float* vr_f = (const float*)d_in[8];
    const float* bf_f = (const float*)d_in[9];
    const float* br_f = (const float*)d_in[10];
    const float* al_f = (const float*)d_in[11];  // alpha (1,)
    float* out_f = (float*)d_out;                // (B,N,3H)

    // ---- workspace carve-out (256B aligned), with lifetime aliasing ----
    char* ws = (char*)d_ws;
    size_t o = 0;
    auto carve = [&](size_t bytes) -> char* {
        char* p = ws + o;
        o += (bytes + 255) & ~(size_t)255;
        return p;
    };
    const size_t nQ = (size_t)PBN * PH;     // 16M elements
    const size_t nS = (size_t)PB * PN * PN; // 33.5M elements
    const size_t nU = (size_t)PBN * PH3;    // 50M elements

    bf16_t* xbf  = (bf16_t*)carve(nQ * 2);            // reused later as Gbf
    bf16_t* WqT  = (bf16_t*)carve((size_t)PH * PD * 2);   // (H,D)
    bf16_t* WkT  = (bf16_t*)carve((size_t)PH * PH * 2);
    bf16_t* WvT  = (bf16_t*)carve((size_t)PH * PH * 2);
    bf16_t* WuT  = (bf16_t*)carve((size_t)PH3 * PH * 2);  // (3H,H)
    bf16_t* Wb   = (bf16_t*)carve((size_t)PH3 * PH * 2);  // W straight (3H,H)
    float*  Qf   = (float*)carve(nQ * 4);             // reused later as Hout
    bf16_t* Qbf  = (bf16_t*)carve(nQ * 2);            // reused later as Hbf
    bf16_t* Kbf  = (bf16_t*)carve(nQ * 2);            // (B*N, H) n-major per batch
    bf16_t* VT   = (bf16_t*)carve(nQ * 2);            // (H, B*N): VT[h][b*N+m]
    float*  S    = (float*)carve(nS * 4);
    bf16_t* Pbf  = (bf16_t*)carve(nS * 2);
    float*  Fa   = (float*)carve(nQ * 4);
    float*  U    = (float*)carve(nU * 4);
    bf16_t* Gbf  = xbf;   // alias: x bf16 dead after Q GEMM
    float*  Hout = Qf;    // alias: Q f32 dead after G
    bf16_t* Hbf  = Qbf;   // alias: Q bf16 dead after scores
    (void)ws_size; (void)in_sizes; (void)n_in; (void)out_size;

    const dim3 blk(256);

    // ---- 1. converts: x straight; Wq/Wk/Wv/Wu transposed; W straight ----
    f32_to_bf16_kernel<<<dim3(4096), blk, 0, stream>>>(x_f, xbf, nQ);
    f32_to_bf16_t_kernel<<<dim3(PH/32,  PD/32),  blk, 0, stream>>>(Wq_f, WqT, PD, PH);
    f32_to_bf16_t_kernel<<<dim3(PH/32,  PH/32),  blk, 0, stream>>>(Wk_f, WkT, PH, PH);
    f32_to_bf16_t_kernel<<<dim3(PH/32,  PH/32),  blk, 0, stream>>>(Wv_f, WvT, PH, PH);
    f32_to_bf16_t_kernel<<<dim3(PH3/32, PH/32),  blk, 0, stream>>>(Wu_f, WuT, PH, PH3);
    f32_to_bf16_kernel<<<dim3(2048), blk, 0, stream>>>(W_f, Wb, (size_t)PH3 * PH);

    // ---- 2. Q = x @ WqT^T  (f32 out; also bf16 copy) ----
    gemm_tn<false,false,false><<<dim3(PBN/BM, PH/BNT, 1), blk, 0, stream>>>(
        xbf, WqT, Qf, nullptr, PBN, PH, PD, PD, PD, PH, 0, 0, 0, 1.0f);
    f32_to_bf16_kernel<<<dim3(4096), blk, 0, stream>>>(Qf, Qbf, nQ);

    // ---- 3. K = Q@Wk (bf16);  V^T = (Q@Wv)^T (bf16, transposed store) ----
    gemm_tn<true,false,false><<<dim3(PBN/BM, PH/BNT, 1), blk, 0, stream>>>(
        Qbf, WkT, Kbf, nullptr, PBN, PH, PH, PH, PH, PH, 0, 0, 0, 1.0f);
    gemm_tn<true,true,false><<<dim3(PBN/BM, PH/BNT, 1), blk, 0, stream>>>(
        Qbf, WvT, VT, nullptr, PBN, PH, PH, PH, PH, /*ldC=*/PBN, 0, 0, 0, 1.0f);

    // ---- 4. S = Q @ K^T / sqrt(H)  (batched over B; K already n-major) ----
    gemm_tn<false,false,false><<<dim3(PN/BM, PN/BNT, PB), blk, 0, stream>>>(
        Qbf, Kbf, S, nullptr, PN, PN, PH, PH, PH, PN,
        (long long)PN * PH, (long long)PN * PH, (long long)PN * PN, 0.03125f);

    // ---- 5. P = softmax(S)  (bf16) ----
    softmax_rows_kernel<<<dim3(PB * PN), blk, 0, stream>>>(S, Pbf, PN);

    // ---- 6. Fa = P @ V  (batched; B = V^T, ldB = B*N, batch offset 2048 cols) ----
    gemm_tn<false,false,false><<<dim3(PN/BM, PH/BNT, PB), blk, 0, stream>>>(
        Pbf, VT, Fa, nullptr, PN, PH, PN, PN, PBN, PH,
        (long long)PN * PN, (long long)PN, (long long)PN * PH, 1.0f);

    // ---- 7. G = bf16(Q + alpha*Fa) ----
    make_g_kernel<<<dim3(4096), blk, 0, stream>>>(Qf, Fa, al_f, Gbf, nQ);

    // ---- 8. U = G @ Wu + bu ----
    gemm_tn<false,false,true><<<dim3(PBN/BM, PH3/BNT, 1), blk, 0, stream>>>(
        Gbf, WuT, U, bu_f, PBN, PH3, PH, PH, PH, PH3, 0, 0, 0, 1.0f);

    // ---- 9. SRU scan -> Hout (f32), then bf16 ----
    sru_scan_kernel<<<dim3((PB * PH) / 256), blk, 0, stream>>>(
        U, x_f, vf_f, vr_f, bf_f, br_f, Hout);
    f32_to_bf16_kernel<<<dim3(4096), blk, 0, stream>>>(Hout, Hbf, nQ);

    // ---- 10. out = H @ W^T  (W already n-major) ----
    gemm_tn<false,false,false><<<dim3(PBN/BM, PH3/BNT, 1), blk, 0, stream>>>(
        Hbf, Wb, out_f, nullptr, PBN, PH3, PH, PH, PH, PH3, 0, 0, 0, 1.0f);
}
